// ConvolutionalLayer_21887153340656
// MI455X (gfx1250) — matrix operand order repaired and checked by reading, not compiled, not run on hardware
//
#include <hip/hip_runtime.h>
#include <stdint.h>

typedef __attribute__((ext_vector_type(16))) __bf16 v16bf;
typedef __attribute__((ext_vector_type(8)))  float  v8f;
typedef __attribute__((ext_vector_type(4)))  int    i32x4;

#define EPSBN 1e-5f
#define AADD(p,v) __hip_atomic_fetch_add((p),(v),__ATOMIC_RELAXED,__HIP_MEMORY_SCOPE_AGENT)
#define LADD(p,v) __hip_atomic_fetch_add((p),(v),__ATOMIC_RELAXED,__HIP_MEMORY_SCOPE_WORKGROUP)

#define AS1 __attribute__((address_space(1)))
#define AS3 __attribute__((address_space(3)))

#if __has_builtin(__builtin_amdgcn_global_load_async_to_lds_b128) && \
    __has_builtin(__builtin_amdgcn_s_wait_asynccnt)
#define USE_ASYNC_LDS 1
#else
#define USE_ASYNC_LDS 0
#endif

__device__ __forceinline__ void async_g2l_b128(const void* g, void* l) {
#if USE_ASYNC_LDS
    __builtin_amdgcn_global_load_async_to_lds_b128(
        (AS1 i32x4*)(g), (AS3 i32x4*)(l), 0, 0);
#else
    *(uint4*)l = *(const uint4*)g;
#endif
}
__device__ __forceinline__ void async_wait() {
#if USE_ASYNC_LDS
    __builtin_amdgcn_s_wait_asynccnt(0);
#endif
}

__device__ __forceinline__ unsigned short f2bf(float f) {
    unsigned int u = __float_as_uint(f);
    u += 0x7fffu + ((u >> 16) & 1u);            // round-to-nearest-even
    return (unsigned short)(u >> 16);
}
__device__ __forceinline__ unsigned int pack2(float a, float b) {
    return (unsigned int)f2bf(a) | ((unsigned int)f2bf(b) << 16);
}

// ---------------- zero workspace region ----------------
__global__ void K_zero(float* __restrict__ p, long n) {
    long i = (long)blockIdx.x * 256 + threadIdx.x;
    if (i < n) p[i] = 0.f;
}

// ---------------- segment_sum(x[src], tgt): one wave per edge ----------------
__global__ __launch_bounds__(256) void K_scatter(
    const float* __restrict__ x, const int* __restrict__ tgt,
    const int* __restrict__ src, float* __restrict__ nbr, int E)
{
    long gid = (long)blockIdx.x * 256 + threadIdx.x;
    long e = gid >> 5;
    int lane = (int)(gid & 31);
    if (e >= E) return;
    int s = src[e], t = tgt[e];
    float4 v = ((const float4*)(x + (size_t)s * 128))[lane];
    float* d = nbr + (size_t)t * 128 + lane * 4;
    AADD(d + 0, v.x); AADD(d + 1, v.y); AADD(d + 2, v.z); AADD(d + 3, v.w);
}

// ---------------- BN1 column stats over h = [nbr[tgt] , x[src]] ----------------
__global__ __launch_bounds__(256) void K_stats1(
    const float* __restrict__ x, const float* __restrict__ nbr,
    const int* __restrict__ tgt, const int* __restrict__ src,
    float* __restrict__ stats1 /* sum[256], ssq[256] */, int E)
{
    __shared__ float ls[512];
    const int tid = threadIdx.x, lane = tid & 31, wv = tid >> 5;
    ls[tid] = 0.f; ls[tid + 256] = 0.f;
    float sa[4] = {0,0,0,0}, qa[4] = {0,0,0,0};
    float sb[4] = {0,0,0,0}, qb[4] = {0,0,0,0};
    long base = (long)blockIdx.x * 256 + wv * 32;
    for (int i = 0; i < 32; ++i) {
        long e = base + i;
        if (e >= E) break;
        int t = tgt[e], s = src[e];
        float4 a = ((const float4*)(nbr + (size_t)t * 128))[lane];
        float4 b = ((const float4*)(x   + (size_t)s * 128))[lane];
        sa[0]+=a.x; qa[0]+=a.x*a.x; sa[1]+=a.y; qa[1]+=a.y*a.y;
        sa[2]+=a.z; qa[2]+=a.z*a.z; sa[3]+=a.w; qa[3]+=a.w*a.w;
        sb[0]+=b.x; qb[0]+=b.x*b.x; sb[1]+=b.y; qb[1]+=b.y*b.y;
        sb[2]+=b.z; qb[2]+=b.z*b.z; sb[3]+=b.w; qb[3]+=b.w*b.w;
    }
    __syncthreads();
    int c = lane * 4;
#pragma unroll
    for (int j = 0; j < 4; ++j) {
        LADD(&ls[c + j],        sa[j]);
        LADD(&ls[128 + c + j],  sb[j]);
        LADD(&ls[256 + c + j],  qa[j]);
        LADD(&ls[384 + c + j],  qb[j]);
    }
    __syncthreads();
    AADD(&stats1[tid],       ls[tid]);
    AADD(&stats1[256 + tid], ls[256 + tid]);
}

// ---------------- fold BN1 into W1/b1, cast to bf16 ----------------
__global__ void K_prep1(const float* __restrict__ W1, const float* __restrict__ b1,
                        const float* __restrict__ g1, const float* __restrict__ be1,
                        const float* __restrict__ stats1,
                        unsigned short* __restrict__ W1bf, float* __restrict__ b1p,
                        float invE)
{
    __shared__ float sc[256], sh[256];
    int tid = threadIdx.x;
    if (tid < 256) {
        float mu  = stats1[tid] * invE;
        float var = stats1[256 + tid] * invE - mu * mu;
        float s   = g1[tid] * rsqrtf(var + EPSBN);
        sc[tid] = s;
        sh[tid] = be1[tid] - mu * s;
    }
    __syncthreads();
    if (tid < 128) {
        float acc = 0.f;
        for (int c = 0; c < 256; ++c) {
            float w = W1[tid * 256 + c];
            W1bf[tid * 256 + c] = f2bf(w * sc[c]);
            acc += w * sh[c];
        }
        b1p[tid] = b1[tid] + acc;
    }
}

__global__ void K_prep2(const float* __restrict__ W2, const float* __restrict__ b2,
                        const float* __restrict__ g2, const float* __restrict__ be2,
                        const float* __restrict__ stats2,
                        unsigned short* __restrict__ W2bf, float* __restrict__ b2p,
                        float invE)
{
    __shared__ float sc[128], sh[128];
    int tid = threadIdx.x;
    if (tid < 128) {
        float mu  = stats2[tid] * invE;
        float var = stats2[128 + tid] * invE - mu * mu;
        float s   = g2[tid] * rsqrtf(var + EPSBN);
        sc[tid] = s;
        sh[tid] = be2[tid] - mu * s;
    }
    __syncthreads();
    if (tid < 128) {
        float acc = 0.f;
        for (int c = 0; c < 128; ++c) {
            float w = W2[tid * 128 + c];
            W2bf[tid * 128 + c] = f2bf(w * sc[c]);
            acc += w * sh[c];
        }
        b2p[tid] = b2[tid] + acc;
    }
}

// ---------------- GEMM1: [128 edges x 256] @ [256 x 128]^T, fused gather +
// BN1-folded weights + bias + ReLU + BN2 column stats, bf16 WMMA ----------------
__global__ __launch_bounds__(256, 1) void K_gemm1(
    const float* __restrict__ x, const float* __restrict__ nbr,
    const int* __restrict__ tgt, const int* __restrict__ src,
    const unsigned short* __restrict__ W1bf, const float* __restrict__ b1p,
    unsigned short* __restrict__ h1, float* __restrict__ stats2, int E)
{
    constexpr int LDA = 136;                    // 128 + 8 pad; 272B rows keep 16B align
    __shared__ __align__(16) unsigned short sA[128 * LDA];
    __shared__ __align__(16) unsigned short sB[128 * LDA];
    __shared__ float sSum[128], sSsq[128];
    const int tid = threadIdx.x, lane = tid & 31, wv = tid >> 5;
    const int half = lane >> 4, lr = lane & 15;
    const long blockBase = (long)blockIdx.x * 128;
    if (tid < 128) { sSum[tid] = 0.f; sSsq[tid] = 0.f; }

    v8f acc[8];
#pragma unroll
    for (int nt = 0; nt < 8; ++nt)
#pragma unroll
        for (int j = 0; j < 8; ++j) acc[nt][j] = 0.f;

    union Frag { v16bf v; uint4 q[2]; };

    for (int ph = 0; ph < 2; ++ph) {            // K split: 0 -> nbr cols, 1 -> x cols
        const int kbase = ph * 128;
        // stage B half (128 out x 128 k) via async LDS DMA (no VGPR round-trip)
#pragma unroll
        for (int i = 0; i < 8; ++i) {
            int c = tid + 256 * i;
            int row = c >> 4, col = (c & 15) * 8;
            async_g2l_b128(&W1bf[row * 256 + kbase + col], &sB[row * LDA + col]);
        }
        // stage A half: gathered rows, f32 -> bf16; each lane packs 8 cols -> b128 store
#pragma unroll
        for (int i = 0; i < 8; ++i) {
            int r = wv * 16 + 2 * i + half;
            long e = blockBase + r; if (e >= E) e = E - 1;
            const float* sp = (ph == 0) ? (nbr + (size_t)tgt[e] * 128)
                                        : (x   + (size_t)src[e] * 128);
            float4 f0 = ((const float4*)sp)[lr * 2];
            float4 f1 = ((const float4*)sp)[lr * 2 + 1];
            uint4 pk;
            pk.x = pack2(f0.x, f0.y); pk.y = pack2(f0.z, f0.w);
            pk.z = pack2(f1.x, f1.y); pk.w = pack2(f1.z, f1.w);
            *(uint4*)&sA[r * LDA + lr * 8] = pk;
        }
        async_wait();
        __syncthreads();
        const int r0 = wv * 16;
#pragma unroll
        for (int kt = 0; kt < 4; ++kt) {
            const int k0 = kt * 32;
            Frag a;                              // ISA 16-bit A layout (16x32)
            {
                int m = r0 + lr;
                int kb = k0 + (half ? 8 : 0);
                a.q[0] = *(const uint4*)&sA[m * LDA + kb];
                a.q[1] = *(const uint4*)&sA[m * LDA + kb + 16];
            }
            Frag b[8];                           // preload all B frags: pipelined DS
#pragma unroll
            for (int nt = 0; nt < 8; ++nt) {
                int n = nt * 16 + lr;
                int kb = k0 + (half ? 8 : 0);
                b[nt].q[0] = *(const uint4*)&sB[n * LDA + kb];
                b[nt].q[1] = *(const uint4*)&sB[n * LDA + kb + 16];
            }
#pragma unroll
            for (int nt = 0; nt < 8; ++nt)       // 8 independent back-to-back WMMAs
                acc[nt] = __builtin_amdgcn_wmma_f32_16x16x32_bf16(
                    false, a.v, false, b[nt].v, (short)0, acc[nt], false, false);
        }
        __syncthreads();
    }
    // epilogue: bias + ReLU, bf16 store, BN2 column stats
    const int mbase = wv * 16 + (half ? 8 : 0);
#pragma unroll
    for (int nt = 0; nt < 8; ++nt) {
        const int n = nt * 16 + lr;
        const float bias = b1p[n];
        float lsum = 0.f, lssq = 0.f;
#pragma unroll
        for (int i = 0; i < 8; ++i) {
            long e = blockBase + mbase + i;
            float v = acc[nt][i] + bias;
            v = v > 0.f ? v : 0.f;
            if (e < E) h1[(size_t)e * 128 + n] = f2bf(v);
            lsum += v; lssq += v * v;
        }
        LADD(&sSum[n], lsum); LADD(&sSsq[n], lssq);
    }
    __syncthreads();
    if (tid < 128) {
        AADD(&stats2[tid],       sSum[tid]);
        AADD(&stats2[128 + tid], sSsq[tid]);
    }
}

// ---------------- GEMM2: [128 edges x 128] @ [128 x 128]^T + bias + ReLU ----------------
__global__ __launch_bounds__(256, 1) void K_gemm2(
    const unsigned short* __restrict__ h1, const unsigned short* __restrict__ W2bf,
    const float* __restrict__ b2p, float* __restrict__ out, int E)
{
    constexpr int LDA = 136;
    __shared__ __align__(16) unsigned short sA[128 * LDA];
    __shared__ __align__(16) unsigned short sB[128 * LDA];
    const int tid = threadIdx.x, lane = tid & 31, wv = tid >> 5;
    const int half = lane >> 4, lr = lane & 15;
    const long blockBase = (long)blockIdx.x * 128;

    // L2 prefetch of the next block's streaming A region (global_prefetch_b8)
    {
        long pe = blockBase + 128;
        if (pe + 128 <= E) {
            const unsigned short* pf = &h1[(size_t)pe * 128];
            __builtin_prefetch(pf + (size_t)tid * 64, 0, 1);
        }
    }

    // stage A (streaming bf16 h1 tile) + B (W2bf) via async LDS DMA
#pragma unroll
    for (int i = 0; i < 8; ++i) {
        int c = tid + 256 * i;
        int row = c >> 4, col = (c & 15) * 8;
        async_g2l_b128(&W2bf[row * 128 + col], &sB[row * LDA + col]);
        long e = blockBase + row; if (e >= E) e = E - 1;
        async_g2l_b128(&h1[(size_t)e * 128 + col], &sA[row * LDA + col]);
    }
    async_wait();
    __syncthreads();

    v8f acc[8];
#pragma unroll
    for (int nt = 0; nt < 8; ++nt)
#pragma unroll
        for (int j = 0; j < 8; ++j) acc[nt][j] = 0.f;

    union Frag { v16bf v; uint4 q[2]; };
    const int r0 = wv * 16;
#pragma unroll
    for (int kt = 0; kt < 4; ++kt) {
        const int k0 = kt * 32;
        Frag a;
        {
            int m = r0 + lr;
            int kb = k0 + (half ? 8 : 0);
            a.q[0] = *(const uint4*)&sA[m * LDA + kb];
            a.q[1] = *(const uint4*)&sA[m * LDA + kb + 16];
        }
        Frag b[8];
#pragma unroll
        for (int nt = 0; nt < 8; ++nt) {
            int n = nt * 16 + lr;
            int kb = k0 + (half ? 8 : 0);
            b[nt].q[0] = *(const uint4*)&sB[n * LDA + kb];
            b[nt].q[1] = *(const uint4*)&sB[n * LDA + kb + 16];
        }
#pragma unroll
        for (int nt = 0; nt < 8; ++nt)
            acc[nt] = __builtin_amdgcn_wmma_f32_16x16x32_bf16(
                false, a.v, false, b[nt].v, (short)0, acc[nt], false, false);
    }
    const int mbase = r0 + (half ? 8 : 0);
#pragma unroll
    for (int nt = 0; nt < 8; ++nt) {
        const int n = nt * 16 + lr;
        const float bias = b2p[n];
#pragma unroll
        for (int i = 0; i < 8; ++i) {
            long e = blockBase + mbase + i;
            if (e < E) {
                float v = acc[nt][i] + bias;
                out[(size_t)e * 128 + n] = v > 0.f ? v : 0.f;
            }
        }
    }
}

extern "C" void kernel_launch(void* const* d_in, const int* in_sizes, int n_in,
                              void* d_out, int out_size, void* d_ws, size_t ws_size,
                              hipStream_t stream)
{
    (void)n_in; (void)out_size; (void)ws_size;
    const float* x      = (const float*)d_in[0];
    const float* gamma1 = (const float*)d_in[1];
    const float* beta1  = (const float*)d_in[2];
    const float* W1     = (const float*)d_in[3];
    const float* b1     = (const float*)d_in[4];
    const float* gamma2 = (const float*)d_in[5];
    const float* beta2  = (const float*)d_in[6];
    const float* W2     = (const float*)d_in[7];
    const float* b2     = (const float*)d_in[8];
    const int*   tgt    = (const int*)d_in[9];
    const int*   srcI   = (const int*)d_in[10];
    const int N = in_sizes[0] / 128;
    const int E = in_sizes[9];

    // workspace layout (f32 units)
    float* ws      = (float*)d_ws;
    float* nbr     = ws;                                  // N*128
    float* stats1  = ws + (size_t)N * 128;                // 512
    float* stats2  = stats1 + 512;                        // 256
    float* b1p     = stats2 + 256;                        // 128
    float* b2p     = b1p + 128;                           // 128
    unsigned short* W1bf = (unsigned short*)(b2p + 128);  // 128*256 bf16
    unsigned short* W2bf = W1bf + 128 * 256;              // 128*128 bf16
    unsigned short* h1   = W2bf + 128 * 128;              // E*128 bf16

    long zn = (long)N * 128 + 768;                        // nbr + stats1 + stats2
    K_zero<<<dim3((unsigned)((zn + 255) / 256)), dim3(256), 0, stream>>>(ws, zn);

    long sthreads = (long)E * 32;
    K_scatter<<<dim3((unsigned)((sthreads + 255) / 256)), dim3(256), 0, stream>>>(
        x, tgt, srcI, nbr, E);

    K_stats1<<<dim3((unsigned)((E + 255) / 256)), dim3(256), 0, stream>>>(
        x, nbr, tgt, srcI, stats1, E);

    K_prep1<<<dim3(1), dim3(256), 0, stream>>>(
        W1, b1, gamma1, beta1, stats1, W1bf, b1p, (float)(1.0 / (double)E));

    const unsigned gb = (unsigned)((E + 127) / 128);
    K_gemm1<<<dim3(gb), dim3(256), 0, stream>>>(
        x, nbr, tgt, srcI, W1bf, b1p, h1, stats2, E);

    K_prep2<<<dim3(1), dim3(128), 0, stream>>>(
        W2, b2, gamma2, beta2, stats2, W2bf, b2p, (float)(1.0 / (double)E));

    K_gemm2<<<dim3(gb), dim3(256), 0, stream>>>(h1, W2bf, b2p, (float*)d_out, E);
}
